// GPTCacheFlowAttention_2207613190843
// MI455X (gfx1250) — matrix operand order repaired and checked by reading, not compiled, not run on hardware
//
#include <hip/hip_runtime.h>

typedef __attribute__((ext_vector_type(16))) _Float16 v16h;
typedef __attribute__((ext_vector_type(8)))  float    v8f;
typedef unsigned int u32x4 __attribute__((ext_vector_type(4)));
typedef int          i32x4 __attribute__((ext_vector_type(4)));
typedef int          i32x8 __attribute__((ext_vector_type(8)));

#define NH 32
#define HS 128
#define BS 16
#define ATTN_SCALE 0.08838834764831845f
#define NEG_INF -1000000000.0f

#if defined(__has_builtin)
#if __has_builtin(__builtin_amdgcn_tensor_load_to_lds) && \
    __has_builtin(__builtin_amdgcn_s_wait_tensorcnt)
#define HAVE_TDM 1
#endif
#endif

#ifdef HAVE_TDM
// ---------------------------------------------------------------------------
// Issue one TDM 2-D tile load (global f16 -> LDS) per CDNA5 D# layout.
// tile_d0 = contiguous elements per row, tile_d1 = rows,
// stride0  = row stride in elements, TDM inserts pad_amount after each row
// so tiles land directly in the bank-padded LDS layout the WMMA frags read.
// ---------------------------------------------------------------------------
__device__ __forceinline__ void tdm_load_2d_f16(unsigned int lds_addr,
                                                const void* gaddr,
                                                unsigned int tile_d0,
                                                unsigned int tile_d1,
                                                unsigned long stride0,
                                                unsigned int pad_interval,
                                                unsigned int pad_amount)
{
    unsigned long ga = (unsigned long)(uintptr_t)gaddr;
    u32x4 g0; i32x8 g1; i32x4 g2 = {}; i32x4 g3 = {};
    g0[0] = 1u;                                        // count=1, user desc
    g0[1] = lds_addr;                                  // LDS byte offset
    g0[2] = (unsigned int)ga;                          // global_addr[31:0]
    g0[3] = (unsigned int)((ga >> 32) & 0x01FFFFFFu)   // global_addr[56:32]
          | (2u << 30);                                // type = 2 ("image")
    const unsigned int td0 = 1u << 20, td1 = 1u << 20; // generous dims (no OOB)
    g1[0] = (int)((1u << 16)                           // data_size = 2 bytes
          | (1u << 20)                                 // pad_enable
          | (pad_interval << 22) | (pad_amount << 25));
    g1[1] = (int)((td0 & 0xFFFFu) << 16);              // tensor_dim0 lo
    g1[2] = (int)((td0 >> 16) | ((td1 & 0xFFFFu) << 16));
    g1[3] = (int)((td1 >> 16) | (tile_d0 << 16));
    g1[4] = (int)tile_d1;                              // tile_dim2 = 0
    g1[5] = (int)(unsigned int)(stride0 & 0xFFFFFFFFul);
    g1[6] = (int)(unsigned int)(stride0 >> 32);        // dim1_stride = 0 (2-D)
    g1[7] = 0;
#if __clang_major__ >= 23
    i32x8 gx = {};
    __builtin_amdgcn_tensor_load_to_lds(g0, g1, g2, g3, gx, 0);
#else
    __builtin_amdgcn_tensor_load_to_lds(g0, g1, g2, g3, 0);
#endif
}
#endif // HAVE_TDM

// ---------------------------------------------------------------------------
// Pre-pass: fp32 -> f16 for the prompt region. K row-major; V transposed to
// [seq][head][dim][tok] so PV B-fragments are contiguous in LDS.
// ---------------------------------------------------------------------------
__global__ void convert_kv_kernel(const float* __restrict__ k,
                                  const float* __restrict__ v,
                                  _Float16* __restrict__ kf,
                                  _Float16* __restrict__ vt,
                                  int NP_, int L)
{
    const long total = (long)NP_ * (NH * HS);
    for (long i = (long)blockIdx.x * blockDim.x + threadIdx.x; i < total;
         i += (long)gridDim.x * blockDim.x) {
        kf[i] = (_Float16)k[i];
        const int tok = (int)(i >> 12);
        const int c   = (int)(i & 4095);
        const int h   = c >> 7, d = c & 127;
        const int seq = tok / L, t = tok - seq * L;
        vt[(((long)seq * NH + h) * HS + d) * L + t] = (_Float16)v[i];
    }
}

// ---------------------------------------------------------------------------
// Prompt flash attention using TDM-staged f16 K / transposed-V tiles.
// One block = 4 waves = 64 Q rows of one (seq, head); keys in chunks of 32.
// ---------------------------------------------------------------------------
__global__ __launch_bounds__(128)
void prompt_attn_tdm_kernel(const float* __restrict__ q,
                            const _Float16* __restrict__ kf,
                            const _Float16* __restrict__ vt,
                            float* __restrict__ out,
                            int L)
{
    const int QT  = L / 64;
    const int bid = blockIdx.x;
    const int qt  = bid % QT;
    const int h   = (bid / QT) % NH;
    const int seq = bid / (QT * NH);

    __shared__ __align__(16) _Float16 sQ[64][HS + 8];   // Q tile (row-major)
    __shared__ __align__(16) _Float16 sK[32][HS + 8];   // K chunk [key][dim]
    __shared__ __align__(16) _Float16 sVt[HS][32 + 8];  // V chunk [dim][key]
    __shared__ __align__(16) _Float16 sP[4][16][32];    // per-wave P relayout

    const int tid  = threadIdx.x;
    const int wave = tid >> 5;
    const int lane = tid & 31;

    const int  qbase     = qt * 64;
    const long seq_tok0  = (long)seq * L;
    const long rowstride = (long)NH * HS;

    // ---- stage Q tile (64 x 128) fp32 -> f16 ----
    for (int e = tid; e < 64 * HS; e += 128) {
        int r = e >> 7, d = e & 127;
        sQ[r][d] = (_Float16)q[(seq_tok0 + qbase + r) * rowstride + h * HS + d];
    }
    __syncthreads();

    // ---- A-fragments of Q (4 K-steps of 32), contiguous k = 16*(lane/16) + i order (as B) ----
    const int m   = lane & 15;
    const int klo = (lane < 16) ? 0 : 16;
    const int khi = klo + 8;
    const int qrowL = qbase + wave * 16;
    v16h afrag[4];
    #pragma unroll
    for (int s = 0; s < 4; ++s) {
        #pragma unroll
        for (int i = 0; i < 8; ++i) {
            afrag[s][i]     = sQ[wave * 16 + m][32 * s + klo + i];
            afrag[s][8 + i] = sQ[wave * 16 + m][32 * s + khi + i];
        }
    }

    const int hi8 = (lane >> 4) * 8;
    const int n   = lane & 15;
    const int kb  = (lane < 16) ? 0 : 16;
    float mrow[8], lrow[8];
    v8f acc[8];
    const v8f vzero = {};
    #pragma unroll
    for (int r = 0; r < 8; ++r) { mrow[r] = -3.0e38f; lrow[r] = 0.0f; }
    #pragma unroll
    for (int hc = 0; hc < 8; ++hc) acc[hc] = vzero;

    const _Float16* kchunk0 = kf + (seq_tok0 * rowstride + h * HS);
    const _Float16* vchunk0 = vt + ((long)seq * NH + h) * HS * L;

    const int nChunks = (qbase + 64) / 32;
    for (int kc = 0; kc < nChunks; ++kc) {
        __syncthreads();                       // retire prior sK/sVt readers
#ifdef HAVE_TDM
        if (tid < 32) {                        // wave 0 drives the TDM
            // K: 32 rows x 128 elems, row stride 4096 elems, pad 16B/row
            tdm_load_2d_f16((unsigned int)(uintptr_t)&sK[0][0],
                            kchunk0 + (long)kc * 32 * rowstride,
                            HS, 32, (unsigned long)rowstride,
                            /*pad_interval(256B)=*/5u, /*pad_amount(16B)=*/3u);
            // V^T: 128 rows x 32 elems, row stride L elems, pad 16B/row
            tdm_load_2d_f16((unsigned int)(uintptr_t)&sVt[0][0],
                            vchunk0 + kc * 32,
                            32, HS, (unsigned long)L,
                            /*pad_interval(64B)=*/3u, /*pad_amount(16B)=*/3u);
            __builtin_amdgcn_s_wait_tensorcnt(0);
        }
#else
        for (int e = tid; e < 32 * HS; e += 128) {     // sync fallback
            int r = e >> 7, d = e & 127;
            sK[r][d] = kchunk0[((long)kc * 32 + r) * rowstride + d];
        }
        for (int e = tid; e < 32 * HS; e += 128) {
            int dd = e >> 5, t = e & 31;
            sVt[dd][t] = vchunk0[(long)dd * L + kc * 32 + t];
        }
#endif
        __syncthreads();

        // ---- S = Q K^T : two 16x16 N-tiles, 4 K-steps each ----
        v8f c0 = vzero, c1 = vzero;
        #pragma unroll
        for (int s = 0; s < 4; ++s) {
            v16h b0, b1;
            #pragma unroll
            for (int j = 0; j < 16; ++j) {
                b0[j] = sK[n][32 * s + kb + j];
                b1[j] = sK[16 + n][32 * s + kb + j];
            }
            c0 = __builtin_amdgcn_wmma_f32_16x16x32_f16(false, afrag[s], false, b0,
                                                        (short)0, c0, false, false);
            c1 = __builtin_amdgcn_wmma_f32_16x16x32_f16(false, afrag[s], false, b1,
                                                        (short)0, c1, false, false);
        }

        // ---- scale + causal mask + online softmax ----
        const int kcol0 = kc * 32 + n;
        const int kcol1 = kc * 32 + 16 + n;
        #pragma unroll
        for (int r = 0; r < 8; ++r) {
            const int qr = qrowL + r + hi8;
            float x0 = c0[r] * ATTN_SCALE; if (kcol0 > qr) x0 = NEG_INF;
            float x1 = c1[r] * ATTN_SCALE; if (kcol1 > qr) x1 = NEG_INF;
            float cm = fmaxf(x0, x1);
            #pragma unroll
            for (int off = 8; off >= 1; off >>= 1)
                cm = fmaxf(cm, __shfl_xor(cm, off, 32));
            const float mnew  = fmaxf(mrow[r], cm);
            const float rescl = __expf(mrow[r] - mnew);
            const float p0 = __expf(x0 - mnew);
            const float p1 = __expf(x1 - mnew);
            float ps = p0 + p1;
            #pragma unroll
            for (int off = 8; off >= 1; off >>= 1)
                ps += __shfl_xor(ps, off, 32);
            lrow[r] = lrow[r] * rescl + ps;
            mrow[r] = mnew;
            #pragma unroll
            for (int hc = 0; hc < 8; ++hc) acc[hc][r] *= rescl;
            sP[wave][r + hi8][n]      = (_Float16)p0;
            sP[wave][r + hi8][16 + n] = (_Float16)p1;
        }

        // ---- O += P V : contiguous B-frags from transposed V ----
        v16h pa;
        #pragma unroll
        for (int i = 0; i < 8; ++i) {
            pa[i]     = sP[wave][m][klo + i];
            pa[8 + i] = sP[wave][m][khi + i];
        }
        #pragma unroll
        for (int hc = 0; hc < 8; ++hc) {
            v16h bv;
            #pragma unroll
            for (int j = 0; j < 16; ++j)
                bv[j] = sVt[hc * 16 + n][kb + j];
            acc[hc] = __builtin_amdgcn_wmma_f32_16x16x32_f16(false, pa, false, bv,
                                                             (short)0, acc[hc], false, false);
        }
    }

    #pragma unroll
    for (int hc = 0; hc < 8; ++hc)
        #pragma unroll
        for (int r = 0; r < 8; ++r) {
            long tok = seq_tok0 + qrowL + r + hi8;
            out[tok * rowstride + h * HS + hc * 16 + n] = acc[hc][r] / lrow[r];
        }
}

// ---------------------------------------------------------------------------
// Fallback prompt attention (reads fp32 directly; used if workspace is too
// small for the f16 pre-pass buffers). Same WMMA structure.
// ---------------------------------------------------------------------------
__global__ __launch_bounds__(128)
void prompt_attn_f32_kernel(const float* __restrict__ q,
                            const float* __restrict__ k,
                            const float* __restrict__ v,
                            float* __restrict__ out,
                            int L)
{
    const int QT  = L / 64;
    const int bid = blockIdx.x;
    const int qt  = bid % QT;
    const int h   = (bid / QT) % NH;
    const int seq = bid / (QT * NH);

    __shared__ __align__(16) _Float16 sQ[64][HS + 8];
    __shared__ __align__(16) _Float16 sK[32][HS + 8];
    __shared__ __align__(16) _Float16 sVt[HS][32 + 8];
    __shared__ __align__(16) _Float16 sP[4][16][32];

    const int tid  = threadIdx.x;
    const int wave = tid >> 5;
    const int lane = tid & 31;

    const int  qbase     = qt * 64;
    const long seq_tok0  = (long)seq * L;
    const long rowstride = (long)NH * HS;

    for (int e = tid; e < 64 * HS; e += 128) {
        int r = e >> 7, d = e & 127;
        sQ[r][d] = (_Float16)q[(seq_tok0 + qbase + r) * rowstride + h * HS + d];
    }
    __syncthreads();

    const int m   = lane & 15;
    const int klo = (lane < 16) ? 0 : 16;
    const int khi = klo + 8;
    const int qrowL = qbase + wave * 16;
    v16h afrag[4];
    #pragma unroll
    for (int s = 0; s < 4; ++s)
        #pragma unroll
        for (int i = 0; i < 8; ++i) {
            afrag[s][i]     = sQ[wave * 16 + m][32 * s + klo + i];
            afrag[s][8 + i] = sQ[wave * 16 + m][32 * s + khi + i];
        }

    const int hi8 = (lane >> 4) * 8;
    const int n   = lane & 15;
    const int kb  = (lane < 16) ? 0 : 16;
    float mrow[8], lrow[8];
    v8f acc[8];
    const v8f vzero = {};
    #pragma unroll
    for (int r = 0; r < 8; ++r) { mrow[r] = -3.0e38f; lrow[r] = 0.0f; }
    #pragma unroll
    for (int hc = 0; hc < 8; ++hc) acc[hc] = vzero;

    const int nChunks = (qbase + 64) / 32;
    for (int kc = 0; kc < nChunks; ++kc) {
        __syncthreads();
        for (int e = tid; e < 32 * HS; e += 128) {
            int r = e >> 7, d = e & 127;
            long tok = seq_tok0 + kc * 32 + r;
            float kv = k[tok * rowstride + h * HS + d];
            float vv = v[tok * rowstride + h * HS + d];
            sK[r][d]  = (_Float16)kv;
            sVt[d][r] = (_Float16)vv;
        }
        __syncthreads();

        v8f c0 = vzero, c1 = vzero;
        #pragma unroll
        for (int s = 0; s < 4; ++s) {
            v16h b0, b1;
            #pragma unroll
            for (int j = 0; j < 16; ++j) {
                b0[j] = sK[n][32 * s + kb + j];
                b1[j] = sK[16 + n][32 * s + kb + j];
            }
            c0 = __builtin_amdgcn_wmma_f32_16x16x32_f16(false, afrag[s], false, b0,
                                                        (short)0, c0, false, false);
            c1 = __builtin_amdgcn_wmma_f32_16x16x32_f16(false, afrag[s], false, b1,
                                                        (short)0, c1, false, false);
        }

        const int kcol0 = kc * 32 + n;
        const int kcol1 = kc * 32 + 16 + n;
        #pragma unroll
        for (int r = 0; r < 8; ++r) {
            const int qr = qrowL + r + hi8;
            float x0 = c0[r] * ATTN_SCALE; if (kcol0 > qr) x0 = NEG_INF;
            float x1 = c1[r] * ATTN_SCALE; if (kcol1 > qr) x1 = NEG_INF;
            float cm = fmaxf(x0, x1);
            #pragma unroll
            for (int off = 8; off >= 1; off >>= 1)
                cm = fmaxf(cm, __shfl_xor(cm, off, 32));
            const float mnew  = fmaxf(mrow[r], cm);
            const float rescl = __expf(mrow[r] - mnew);
            const float p0 = __expf(x0 - mnew);
            const float p1 = __expf(x1 - mnew);
            float ps = p0 + p1;
            #pragma unroll
            for (int off = 8; off >= 1; off >>= 1)
                ps += __shfl_xor(ps, off, 32);
            lrow[r] = lrow[r] * rescl + ps;
            mrow[r] = mnew;
            #pragma unroll
            for (int hc = 0; hc < 8; ++hc) acc[hc][r] *= rescl;
            sP[wave][r + hi8][n]      = (_Float16)p0;
            sP[wave][r + hi8][16 + n] = (_Float16)p1;
        }

        v16h pa;
        #pragma unroll
        for (int i = 0; i < 8; ++i) {
            pa[i]     = sP[wave][m][klo + i];
            pa[8 + i] = sP[wave][m][khi + i];
        }
        #pragma unroll
        for (int hc = 0; hc < 8; ++hc) {
            v16h bv;
            #pragma unroll
            for (int j = 0; j < 16; ++j)
                bv[j] = sVt[hc * 16 + n][kb + j];
            acc[hc] = __builtin_amdgcn_wmma_f32_16x16x32_f16(false, pa, false, bv,
                                                             (short)0, acc[hc], false, false);
        }
    }

    #pragma unroll
    for (int hc = 0; hc < 8; ++hc)
        #pragma unroll
        for (int r = 0; r < 8; ++r) {
            long tok = seq_tok0 + qrowL + r + hi8;
            out[tok * rowstride + h * HS + hc * 16 + n] = acc[hc][r] / lrow[r];
        }
}

// ---------------------------------------------------------------------------
// Inverse slot map (KV-cache scatter without mutating input caches).
// ---------------------------------------------------------------------------
__global__ void inv_init_kernel(int* inv, int nslots)
{
    int i = blockIdx.x * blockDim.x + threadIdx.x;
    if (i < nslots) inv[i] = -1;
}

__global__ void inv_scatter_kernel(const int* __restrict__ slot_mapping,
                                   int* inv, int NT)
{
    int t = blockIdx.x * blockDim.x + threadIdx.x;
    if (t < NT) atomicMax(&inv[slot_mapping[t]], t);
}

// ---------------------------------------------------------------------------
// Decode (paged) attention: one block per (g, head). GEMV-shaped -> VALU.
// ---------------------------------------------------------------------------
__global__ __launch_bounds__(128)
void decode_attn_kernel(const float* __restrict__ q,
                        const float* __restrict__ k,
                        const float* __restrict__ v,
                        const float* __restrict__ kcache,
                        const float* __restrict__ vcache,
                        const int* __restrict__ bt,
                        const int* __restrict__ ctxlens,
                        const int* __restrict__ inv,
                        float* __restrict__ out,
                        int NP_, int MB)
{
    const int g   = blockIdx.x / NH;
    const int h   = blockIdx.x % NH;
    const int tid = threadIdx.x;
    const int CTX = MB * BS;
    const long rowstride = (long)NH * HS;

    __shared__ float sq[HS];
    __shared__ float sc[512];
    __shared__ float red[128];

    const int  ctx  = ctxlens[g];
    const long qoff = (long)(NP_ + g) * rowstride + h * HS;
    if (tid < HS) sq[tid] = q[qoff + tid];
    __syncthreads();

    for (int kk = tid; kk < CTX; kk += 128) {
        float s = NEG_INF;
        if (kk < ctx) {
            const int b   = bt[g * MB + kk / BS];
            const int off = kk & (BS - 1);
            const int t   = inv[b * BS + off];
            float a = 0.0f;
            if (t >= 0) {
                const float* kp = k + (long)t * rowstride + h * HS;
                for (int d = 0; d < HS; ++d) a += sq[d] * kp[d];
            } else {
                const float* kp = kcache + ((long)b * NH + h) * HS * BS + off;
                for (int d = 0; d < HS; ++d) a += sq[d] * kp[d * BS];
            }
            s = a * ATTN_SCALE;
        }
        sc[kk] = s;
    }
    __syncthreads();

    float lm = -3.0e38f;
    for (int kk = tid; kk < CTX; kk += 128) lm = fmaxf(lm, sc[kk]);
    red[tid] = lm; __syncthreads();
    for (int s2 = 64; s2 > 0; s2 >>= 1) {
        if (tid < s2) red[tid] = fmaxf(red[tid], red[tid + s2]);
        __syncthreads();
    }
    const float mx = red[0]; __syncthreads();

    float ls = 0.0f;
    for (int kk = tid; kk < CTX; kk += 128) {
        float p = __expf(sc[kk] - mx);
        sc[kk] = p;
        ls += p;
    }
    red[tid] = ls; __syncthreads();
    for (int s2 = 64; s2 > 0; s2 >>= 1) {
        if (tid < s2) red[tid] += red[tid + s2];
        __syncthreads();
    }
    const float sum = red[0]; __syncthreads();

    if (tid < HS) {
        const int d = tid;
        float o = 0.0f;
        for (int kk = 0; kk < CTX; ++kk) {
            const float p   = sc[kk];
            const int   b   = bt[g * MB + kk / BS];
            const int   off = kk & (BS - 1);
            const int   t   = inv[b * BS + off];
            const float vv  = (t >= 0)
                ? v[(long)t * rowstride + h * HS + d]
                : vcache[(((long)b * NH + h) * HS + d) * BS + off];
            o += p * vv;
        }
        out[qoff + d] = o / sum;
    }
}

// ---------------------------------------------------------------------------
extern "C" void kernel_launch(void* const* d_in, const int* in_sizes, int n_in,
                              void* d_out, int out_size, void* d_ws, size_t ws_size,
                              hipStream_t stream)
{
    const float* q       = (const float*)d_in[0];
    const float* k       = (const float*)d_in[1];
    const float* v       = (const float*)d_in[2];
    const float* kcache  = (const float*)d_in[3];
    const float* vcache  = (const float*)d_in[4];
    const int*   bt      = (const int*)d_in[5];
    const int*   ctxlens = (const int*)d_in[6];
    const int*   slotmap = (const int*)d_in[7];
    float*       out     = (float*)d_out;

    const int NHHS = NH * HS;                       // 4096
    const int NT = in_sizes[0] / NHHS;              // 2112
    const int G  = in_sizes[6];                     // 64
    const int MB = in_sizes[5] / G;                 // 32
    const int NP = NT - G;                          // 2048
    const int NB = in_sizes[3] / (NHHS * BS);       // 1024
    const int NS = 4;                               // fixed by reference setup
    const int L  = NP / NS;                         // 512
    const int nslots = NB * BS;                     // 16384

    int* inv = (int*)d_ws;
    const size_t inv_bytes = (((size_t)nslots * 4) + 255) & ~(size_t)255;
    const size_t kf_bytes  = (size_t)NP * NHHS * sizeof(_Float16);
    const size_t need      = inv_bytes + 2 * kf_bytes;

    inv_init_kernel<<<(nslots + 255) / 256, 256, 0, stream>>>(inv, nslots);
    inv_scatter_kernel<<<(NT + 255) / 256, 256, 0, stream>>>(slotmap, inv, NT);

    if (ws_size >= need) {
        _Float16* kf = (_Float16*)((char*)d_ws + inv_bytes);
        _Float16* vt = (_Float16*)((char*)d_ws + inv_bytes + kf_bytes);
        convert_kv_kernel<<<2048, 256, 0, stream>>>(k, v, kf, vt, NP, L);
        prompt_attn_tdm_kernel<<<NS * NH * (L / 64), 128, 0, stream>>>(q, kf, vt, out, L);
    } else {
        prompt_attn_f32_kernel<<<NS * NH * (L / 64), 128, 0, stream>>>(q, k, v, out, L);
    }

    decode_attn_kernel<<<G * NH, 128, 0, stream>>>(q, k, v, kcache, vcache,
                                                   bt, ctxlens, inv, out, NP, MB);
}
